// Explainer_48619029791202
// MI455X (gfx1250) — matrix operand order repaired and checked by reading, not compiled, not run on hardware
//
#include <hip/hip_runtime.h>

#define N_NODES 50000
#define N_EDGES 400000
#define IN_C 128
#define HID 256
#define EDGE_D 32

typedef __attribute__((ext_vector_type(16))) __bf16 v16bf;
typedef __attribute__((ext_vector_type(8)))  float v8f;

union ABFrag { v16bf v; unsigned int u[8]; };

__device__ __forceinline__ unsigned short f2bf(float f) {
    unsigned int u = __float_as_uint(f);
    unsigned int r = u + 0x7FFFu + ((u >> 16) & 1u);   // round-to-nearest-even
    return (unsigned short)(r >> 16);
}

// ---- fragment helpers ----

// A: 16-bit 16x32 layout; row = lane&15, K-half by lane>=16 (kbA in {0,8})
template <int K, int MT>
__device__ __forceinline__ void loadA_frags(const unsigned int* __restrict__ sA32,
                                            ABFrag (&a)[MT], int arow, int kbA, int kk) {
#pragma unroll
    for (int mt = 0; mt < MT; ++mt) {
        int base = ((mt * 16 + arow) * K + kk + 2 * kbA) >> 1;
#pragma unroll
        for (int v = 0; v < 4; ++v) {
            a[mt].u[v]     = sA32[base + v];
            a[mt].u[v + 4] = sA32[base + 4 + v];
        }
    }
}

// B: col = colrow + t*16, contiguous K pairs from N-major (transposed) weights
template <int K>
__device__ __forceinline__ void loadB_frags(const unsigned int* __restrict__ wt32,
                                            ABFrag (&b)[4], int colrow, int kbB, int kk) {
#pragma unroll
    for (int t = 0; t < 4; ++t) {
        int base = ((colrow + t * 16) * K + kk + kbB) >> 1;
#pragma unroll
        for (int v = 0; v < 8; ++v) b[t].u[v] = wt32[base + v];
    }
}

template <int MT>
__device__ __forceinline__ void do_wmma(ABFrag (&a)[MT], ABFrag (&b)[4], v8f (&acc)[MT][4]) {
#pragma unroll
    for (int t = 0; t < 4; ++t)
#pragma unroll
        for (int mt = 0; mt < MT; ++mt)
            acc[mt][t] = __builtin_amdgcn_wmma_f32_16x16x32_bf16(
                false, a[mt].v, false, b[t].v, (short)0, acc[mt][t], false, false);
}

// Software-pipelined K loop: double-buffered A (LDS) and B (global) fragments.
template <int K, int MT>
__device__ __forceinline__ void gemm_pipeline(const unsigned int* __restrict__ sA32,
                                              const unsigned int* __restrict__ wt32,
                                              int arow, int kbA, int colrow, int kbB,
                                              v8f (&acc)[MT][4]) {
    ABFrag a0[MT], a1[MT], b0[4], b1[4];
    loadA_frags<K, MT>(sA32, a0, arow, kbA, 0);
    loadB_frags<K>(wt32, b0, colrow, kbB, 0);
    int kk = 0;
    for (; kk + 64 <= K; kk += 64) {
        loadA_frags<K, MT>(sA32, a1, arow, kbA, kk + 32);
        loadB_frags<K>(wt32, b1, colrow, kbB, kk + 32);
        do_wmma<MT>(a0, b0, acc);
        if (kk + 64 < K) {                       // uniform branch
            loadA_frags<K, MT>(sA32, a0, arow, kbA, kk + 64);
            loadB_frags<K>(wt32, b0, colrow, kbB, kk + 64);
        }
        do_wmma<MT>(a1, b1, acc);
    }
    if (kk < K) do_wmma<MT>(a0, b0, acc);        // odd K/32 tail
}

// ---------------- utility kernels ----------------

__global__ void cvt_bf16_kernel(const float* __restrict__ in,
                                unsigned short* __restrict__ out, int n) {
    int i = blockIdx.x * 256 + threadIdx.x;
    if (i < n) out[i] = f2bf(in[i]);
}

// w: [K,256] fp32 (row-major)  ->  wt: [256,K] bf16 (N-major / transposed)
__global__ void cvt_wT_kernel(const float* __restrict__ w,
                              unsigned short* __restrict__ wt, int K) {
    int i = blockIdx.x * 256 + threadIdx.x;
    if (i < K * 256) {
        int k = i / 256, n = i % 256;
        wt[n * K + k] = f2bf(w[i]);
    }
}

__global__ void zero_kernel(float* __restrict__ p, int n) {
    int i = blockIdx.x * 256 + threadIdx.x;
    if (i < n) p[i] = 0.0f;
}

__global__ void degree_kernel(const int* __restrict__ dst,
                              float* __restrict__ deg, int n) {
    int i = blockIdx.x * 256 + threadIdx.x;
    if (i < n) atomicAdd(&deg[dst[i]], 1.0f);
}

// ---------------- fused edge MLP + scatter-add ----------------
// block = 128 thr (4 waves); tile = 64 edges x 256 outs; K = 2H+32
template <int K, int H>
__global__ __launch_bounds__(128) void edge_gemm_kernel(
    const unsigned short* __restrict__ xf,   // [N_NODES,H] bf16
    const unsigned short* __restrict__ ea,   // [N_EDGES,32] bf16
    const int* __restrict__ src, const int* __restrict__ dst,
    const unsigned short* __restrict__ wt,   // [256,K] bf16 transposed
    const float* __restrict__ bias,          // [256]
    float* __restrict__ agg)                 // [N_NODES,256] f32
{
    constexpr int MT = 4;
    __shared__ unsigned short sA[64 * K];
    __shared__ int sDst[64];
    const int tid = threadIdx.x;

    // ---- gather: 2 threads per edge-row, branch-free region loops ----
    const int row = tid >> 1, lin = tid & 1;
    const int e = blockIdx.x * 64 + row;
    const int se = src[e], de = dst[e];
    if (lin == 0) sDst[row] = de;
    const unsigned short* srow = xf + (size_t)se * H;
    const unsigned short* drow = xf + (size_t)de * H;
    const unsigned short* erow = ea + (size_t)e * EDGE_D;
    unsigned short* ap = sA + row * K;
#pragma unroll
    for (int c = lin; c < H / 8; c += 2)
        *(uint4*)(ap + c * 8) = *(const uint4*)(srow + c * 8);
#pragma unroll
    for (int c = lin; c < H / 8; c += 2)
        *(uint4*)(ap + H + c * 8) = *(const uint4*)(drow + c * 8);
#pragma unroll
    for (int c = lin; c < EDGE_D / 8; c += 2)
        *(uint4*)(ap + 2 * H + c * 8) = *(const uint4*)(erow + c * 8);
    __syncthreads();

    // ---- WMMA phase ----
    const int wave = tid >> 5, lane = tid & 31;
    const int arow = lane & 15;
    const int kbA = (lane < 16) ? 0 : 8;
    const int kbB = (lane < 16) ? 0 : 16;
    const int colrow = wave * 64 + (lane & 15);

    v8f acc[MT][4] = {};
    gemm_pipeline<K, MT>((const unsigned int*)sA, (const unsigned int*)wt,
                         arow, kbA, colrow, kbB, acc);

    // ---- epilogue: bias + relu + scatter-add ----
#pragma unroll
    for (int t = 0; t < 4; ++t) {
        int col = colrow + t * 16;
        float bv = bias[col];
#pragma unroll
        for (int mt = 0; mt < MT; ++mt) {
#pragma unroll
            for (int r = 0; r < 8; ++r) {
                int m = mt * 16 + r + ((lane >> 4) << 3);
                float val = fmaxf(acc[mt][t][r] + bv, 0.0f);
                atomicAdd(&agg[(size_t)sDst[m] * HID + col], val);
            }
        }
    }
}

// ---------------- node MLP: relu(W [h, agg/deg] + b) ----------------
template <int HIN>
__global__ __launch_bounds__(128) void node_gemm_kernel(
    const unsigned short* __restrict__ hin,  // [N_NODES,HIN] bf16
    const float* __restrict__ agg,           // [N_NODES,256] f32
    const float* __restrict__ deg,           // [N_NODES]
    const unsigned short* __restrict__ wt,   // [256,K] bf16 transposed
    const float* __restrict__ bias,
    unsigned short* __restrict__ hout)       // [N_NODES,256] bf16
{
    constexpr int K = HIN + HID;
    constexpr int MT = 4;
    __shared__ unsigned short sA[64 * K];
    const int tid = threadIdx.x;

    const int row = tid >> 1, lin = tid & 1;
    const int node0 = blockIdx.x * 64 + row;
    const int node = (node0 < N_NODES) ? node0 : (N_NODES - 1);   // clamp tail
    const float rd = 1.0f / fmaxf(deg[node], 1.0f);
    const unsigned short* hrow = hin + (size_t)node * HIN;
    const float* grow = agg + (size_t)node * HID;
    unsigned short* ap = sA + row * K;
#pragma unroll
    for (int c = lin; c < HIN / 8; c += 2)
        *(uint4*)(ap + c * 8) = *(const uint4*)(hrow + c * 8);
#pragma unroll
    for (int c = lin; c < HID / 8; c += 2) {
        float4 f0 = *(const float4*)(grow + c * 8);
        float4 f1 = *(const float4*)(grow + c * 8 + 4);
        unsigned short* p = ap + HIN + c * 8;
        p[0] = f2bf(f0.x * rd); p[1] = f2bf(f0.y * rd);
        p[2] = f2bf(f0.z * rd); p[3] = f2bf(f0.w * rd);
        p[4] = f2bf(f1.x * rd); p[5] = f2bf(f1.y * rd);
        p[6] = f2bf(f1.z * rd); p[7] = f2bf(f1.w * rd);
    }
    __syncthreads();

    const int wave = tid >> 5, lane = tid & 31;
    const int arow = lane & 15;
    const int kbA = (lane < 16) ? 0 : 8;
    const int kbB = (lane < 16) ? 0 : 16;
    const int colrow = wave * 64 + (lane & 15);

    v8f acc[MT][4] = {};
    gemm_pipeline<K, MT>((const unsigned int*)sA, (const unsigned int*)wt,
                         arow, kbA, colrow, kbB, acc);

#pragma unroll
    for (int t = 0; t < 4; ++t) {
        int col = colrow + t * 16;
        float bv = bias[col];
#pragma unroll
        for (int mt = 0; mt < MT; ++mt) {
#pragma unroll
            for (int r = 0; r < 8; ++r) {
                int m = mt * 16 + r + ((lane >> 4) << 3);
                int n = blockIdx.x * 64 + m;
                if (n < N_NODES) {
                    float val = fmaxf(acc[mt][t][r] + bv, 0.0f);
                    hout[(size_t)n * HID + col] = f2bf(val);
                }
            }
        }
    }
}

// ---------------- head: relu(h@h1+b1) @ h2 + b2 ----------------
__global__ __launch_bounds__(128) void head_kernel(
    const unsigned short* __restrict__ hin,  // [N,256] bf16
    const unsigned short* __restrict__ w1t,  // [256,256] bf16 transposed
    const float* __restrict__ b1,
    const float* __restrict__ w2,            // [256]
    const float* __restrict__ b2,            // [1]
    float* __restrict__ out)                 // [N]
{
    constexpr int K = 256;
    constexpr int MT = 4;
    __shared__ unsigned short sA[64 * K];
    __shared__ float sH[64 * 8];             // per-row sums (one per row, first 64 used)
    const int tid = threadIdx.x;

    const int row = tid >> 1, lin = tid & 1;
    const int node0 = blockIdx.x * 64 + row;
    const int node = (node0 < N_NODES) ? node0 : (N_NODES - 1);
    const unsigned short* hrow = hin + (size_t)node * HID;
    unsigned short* ap = sA + row * K;
#pragma unroll
    for (int c = lin; c < K / 8; c += 2)
        *(uint4*)(ap + c * 8) = *(const uint4*)(hrow + c * 8);
    __syncthreads();

    const int wave = tid >> 5, lane = tid & 31;
    const int arow = lane & 15;
    const int kbA = (lane < 16) ? 0 : 8;
    const int kbB = (lane < 16) ? 0 : 16;
    const int colrow = wave * 64 + (lane & 15);

    v8f acc[MT][4] = {};
    gemm_pipeline<K, MT>((const unsigned int*)sA, (const unsigned int*)w1t,
                         arow, kbA, colrow, kbB, acc);

    // per-lane partial dot: relu(h1-out) * w2, summed over this lane's 4 cols
    float part[MT];
#pragma unroll
    for (int mt = 0; mt < MT; ++mt) part[mt * 2 + 0] = 0.0f, part[mt * 2 + 1] = 0.0f;
    // (re-init cleanly below)
    float ps[MT][8];
#pragma unroll
    for (int mt = 0; mt < MT; ++mt)
#pragma unroll
        for (int r = 0; r < 8; ++r) ps[mt][r] = 0.0f;
#pragma unroll
    for (int t = 0; t < 4; ++t) {
        int col = colrow + t * 16;
        float bv = b1[col];
        float wv = w2[col];
#pragma unroll
        for (int mt = 0; mt < MT; ++mt)
#pragma unroll
            for (int r = 0; r < 8; ++r)
                ps[mt][r] += fmaxf(acc[mt][t][r] + bv, 0.0f) * wv;
    }
    // each (row m) has 64 partials spread over 64 lanes' ps (16 lanes x 4 waves); reduce via LDS.
    // lane contributes ps[mt][r] to row m = mt*16 + r + 8*(lane>=16), slot = lane&7? 
    // Simpler: atomically accumulate into sH[m] after zeroing.
    float* sAcc = (float*)sH;
    if (tid < 64) sAcc[tid] = 0.0f;
    __syncthreads();
#pragma unroll
    for (int mt = 0; mt < MT; ++mt)
#pragma unroll
        for (int r = 0; r < 8; ++r) {
            int m = mt * 16 + r + ((lane >> 4) << 3);
            atomicAdd(&sAcc[m], ps[mt][r]);
        }
    __syncthreads();
    if (tid < 64) {
        int n = blockIdx.x * 64 + tid;
        if (n < N_NODES) out[n] = sAcc[tid] + b2[0];
    }
}

// ---------------- launch ----------------

extern "C" void kernel_launch(void* const* d_in, const int* in_sizes, int n_in,
                              void* d_out, int out_size, void* d_ws, size_t ws_size,
                              hipStream_t stream) {
    const float* x         = (const float*)d_in[0];
    const int*   edge_idx  = (const int*)d_in[1];
    const float* edge_attr = (const float*)d_in[2];
    const float* e0_w = (const float*)d_in[3];
    const float* e0_b = (const float*)d_in[4];
    const float* n0_w = (const float*)d_in[5];
    const float* n0_b = (const float*)d_in[6];
    const float* e_w  = (const float*)d_in[7];
    const float* e_b  = (const float*)d_in[8];
    const float* n_w  = (const float*)d_in[9];
    const float* n_b  = (const float*)d_in[10];
    const float* h1_w = (const float*)d_in[11];
    const float* h1_b = (const float*)d_in[12];
    const float* h2_w = (const float*)d_in[13];
    const float* h2_b = (const float*)d_in[14];
    float* out = (float*)d_out;

    const int* src = edge_idx;
    const int* dst = edge_idx + N_EDGES;

    char* w = (char*)d_ws;
    auto carve = [&](size_t bytes) {
        char* p = w;
        w += (bytes + 255) & ~(size_t)255;
        return p;
    };
    unsigned short* x_bf  = (unsigned short*)carve((size_t)N_NODES * IN_C * 2);
    unsigned short* ea_bf = (unsigned short*)carve((size_t)N_EDGES * EDGE_D * 2);
    unsigned short* hA    = (unsigned short*)carve((size_t)N_NODES * HID * 2);
    unsigned short* hB    = (unsigned short*)carve((size_t)N_NODES * HID * 2);
    float* agg = (float*)carve((size_t)N_NODES * HID * 4);
    float* deg = (float*)carve((size_t)N_NODES * 4);
    unsigned short* e0wt = (unsigned short*)carve((size_t)288 * 256 * 2);
    unsigned short* n0wt = (unsigned short*)carve((size_t)384 * 256 * 2);
    unsigned short* e1wt = (unsigned short*)carve((size_t)544 * 256 * 2);
    unsigned short* e2wt = (unsigned short*)carve((size_t)544 * 256 * 2);
    unsigned short* n1wt = (unsigned short*)carve((size_t)512 * 256 * 2);
    unsigned short* n2wt = (unsigned short*)carve((size_t)512 * 256 * 2);
    unsigned short* h1wt = (unsigned short*)carve((size_t)256 * 256 * 2);

    auto gcdiv = [](int n, int b) { return (n + b - 1) / b; };

    cvt_bf16_kernel<<<gcdiv(N_NODES * IN_C, 256), 256, 0, stream>>>(x, x_bf, N_NODES * IN_C);
    cvt_bf16_kernel<<<gcdiv(N_EDGES * EDGE_D, 256), 256, 0, stream>>>(edge_attr, ea_bf, N_EDGES * EDGE_D);
    cvt_wT_kernel<<<gcdiv(288 * 256, 256), 256, 0, stream>>>(e0_w, e0wt, 288);
    cvt_wT_kernel<<<gcdiv(384 * 256, 256), 256, 0, stream>>>(n0_w, n0wt, 384);
    cvt_wT_kernel<<<gcdiv(544 * 256, 256), 256, 0, stream>>>(e_w, e1wt, 544);
    cvt_wT_kernel<<<gcdiv(544 * 256, 256), 256, 0, stream>>>(e_w + 544 * 256, e2wt, 544);
    cvt_wT_kernel<<<gcdiv(512 * 256, 256), 256, 0, stream>>>(n_w, n1wt, 512);
    cvt_wT_kernel<<<gcdiv(512 * 256, 256), 256, 0, stream>>>(n_w + 512 * 256, n2wt, 512);
    cvt_wT_kernel<<<gcdiv(256 * 256, 256), 256, 0, stream>>>(h1_w, h1wt, 256);

    zero_kernel<<<gcdiv(N_NODES, 256), 256, 0, stream>>>(deg, N_NODES);
    degree_kernel<<<gcdiv(N_EDGES, 256), 256, 0, stream>>>(dst, deg, N_EDGES);

    const int NAGG = N_NODES * HID;
    const int egrid = N_EDGES / 64;          // 6250 exact
    const int ngrid = gcdiv(N_NODES, 64);    // 782 (tail-guarded)

    // layer 0
    zero_kernel<<<gcdiv(NAGG, 256), 256, 0, stream>>>(agg, NAGG);
    edge_gemm_kernel<288, 128><<<egrid, 128, 0, stream>>>(x_bf, ea_bf, src, dst, e0wt, e0_b, agg);
    node_gemm_kernel<128><<<ngrid, 128, 0, stream>>>(x_bf, agg, deg, n0wt, n0_b, hA);

    // layer 1
    zero_kernel<<<gcdiv(NAGG, 256), 256, 0, stream>>>(agg, NAGG);
    edge_gemm_kernel<544, 256><<<egrid, 128, 0, stream>>>(hA, ea_bf, src, dst, e1wt, e_b, agg);
    node_gemm_kernel<256><<<ngrid, 128, 0, stream>>>(hA, agg, deg, n1wt, n_b, hB);

    // layer 2
    zero_kernel<<<gcdiv(NAGG, 256), 256, 0, stream>>>(agg, NAGG);
    edge_gemm_kernel<544, 256><<<egrid, 128, 0, stream>>>(hB, ea_bf, src, dst, e2wt, e_b + 256, agg);
    node_gemm_kernel<256><<<ngrid, 128, 0, stream>>>(hB, agg, deg, n2wt, n_b + 256, hA);

    // head
    head_kernel<<<ngrid, 128, 0, stream>>>(hA, h1wt, h1_b, h2_w, h2_b, out);
}
